// MultiHeadAttention_61125974557465
// MI455X (gfx1250) — matrix operand order repaired and checked by reading, not compiled, not run on hardware
//
#include <hip/hip_runtime.h>
#include <hip/hip_bf16.h>
#include <stdint.h>

typedef _Float16 half_t;
typedef __attribute__((ext_vector_type(16))) _Float16 v16h;
typedef __attribute__((ext_vector_type(8)))  _Float16 v8h;
typedef __attribute__((ext_vector_type(8)))  float    v8f;

#define H_S   1024
#define N_H   16
#define P_S   64
#define SEQ   2048
#define BATCH 4
#define MTOT  (BATCH * SEQ)   // 8192

// ---------------------------------------------------------------------------
// WMMA helper (CDNA5 16x16x32 f16 -> f32)
// ---------------------------------------------------------------------------
__device__ __forceinline__ v8f wmma32(v16h a, v16h b, v8f c) {
    return __builtin_amdgcn_wmma_f32_16x16x32_f16(
        /*neg_a=*/false, a, /*neg_b=*/false, b,
        /*c_mod=*/(short)0, c, /*reuse_a=*/false, /*reuse_b=*/false);
}

// A-fragment (16x32, row-major source).
// lanes 0-15: M=lane,   halves 0..15 = K k0..k0+15
// lanes16-31: M=lane-16,halves 0..15 = K k0+16..k0+31
__device__ __forceinline__ v16h load_a_frag(const half_t* __restrict__ row_base,
                                            int k0, int hi) {
    const half_t* p = row_base + k0 + hi * 16;
    v8h lo = *(const v8h*)(p);
    v8h hh = *(const v8h*)(p + 8);
    return __builtin_shufflevector(lo, hh, 0,1,2,3,4,5,6,7,8,9,10,11,12,13,14,15);
}

// ---------------------------------------------------------------------------
// CDNA5 async global->LDS copy (ASYNCcnt-tracked), inline asm per
// CDNA5 ISA.  lds_off = byte offset within workgroup LDS
// (low 32 bits of the flat shared address; LDS aperture maps addr[31:0]).
// ---------------------------------------------------------------------------
__device__ __forceinline__ unsigned lds_off(const void* p) {
    return (unsigned)(uintptr_t)p;
}
__device__ __forceinline__ void async_ld_b128(unsigned lds, const half_t* g) {
    asm volatile("global_load_async_to_lds_b128 %0, %1, off"
                 :: "v"(lds), "v"(g) : "memory");
}
#define WAIT_ASYNC(n) asm volatile("s_wait_asynccnt " #n ::: "memory")

// Row-wise reductions across the 16 lanes holding one C-fragment row.
__device__ __forceinline__ float rowmax16(float v) {
    v = fmaxf(v, __shfl_xor(v, 1, 16));
    v = fmaxf(v, __shfl_xor(v, 2, 16));
    v = fmaxf(v, __shfl_xor(v, 4, 16));
    v = fmaxf(v, __shfl_xor(v, 8, 16));
    return v;
}
__device__ __forceinline__ float rowsum16(float v) {
    v += __shfl_xor(v, 1, 16);
    v += __shfl_xor(v, 2, 16);
    v += __shfl_xor(v, 4, 16);
    v += __shfl_xor(v, 8, 16);
    return v;
}

// ---------------------------------------------------------------------------
// Kernel 1: LayerNorm (fp32 in) -> f16 out.  One block per 1024-wide row.
// ---------------------------------------------------------------------------
__global__ __launch_bounds__(256) void layernorm_to_f16(
    const float* __restrict__ x, const float* __restrict__ gamma,
    const float* __restrict__ beta, half_t* __restrict__ y) {
    const int row  = blockIdx.x;
    const int t    = threadIdx.x;
    const int lane = t & 31, wave = t >> 5;
    const float4* xr = (const float4*)(x + (size_t)row * H_S);
    float4 v = xr[t];
    float s  = v.x + v.y + v.z + v.w;
    float ss = v.x * v.x + v.y * v.y + v.z * v.z + v.w * v.w;
#pragma unroll
    for (int m = 1; m < 32; m <<= 1) {
        s  += __shfl_xor(s,  m, 32);
        ss += __shfl_xor(ss, m, 32);
    }
    __shared__ float redS[8], redQ[8];
    if (lane == 0) { redS[wave] = s; redQ[wave] = ss; }
    __syncthreads();
    float ts = 0.f, tq = 0.f;
#pragma unroll
    for (int w = 0; w < 8; w++) { ts += redS[w]; tq += redQ[w]; }
    const float mu  = ts * (1.0f / H_S);
    const float var = tq * (1.0f / H_S) - mu * mu;
    const float rs  = rsqrtf(var + 1e-5f);
    float4 g = ((const float4*)gamma)[t];
    float4 b = ((const float4*)beta)[t];
    half_t* yr = y + (size_t)row * H_S + t * 4;
    yr[0] = (half_t)((v.x - mu) * rs * g.x + b.x);
    yr[1] = (half_t)((v.y - mu) * rs * g.y + b.y);
    yr[2] = (half_t)((v.z - mu) * rs * g.z + b.z);
    yr[3] = (half_t)((v.w - mu) * rs * g.w + b.w);
}

// ---------------------------------------------------------------------------
// Kernel 2: fp32 -> f16 weight conversion
// ---------------------------------------------------------------------------
__global__ __launch_bounds__(256) void cvt_f32_to_f16(
    const float* __restrict__ in, half_t* __restrict__ out, int n) {
    int i = (blockIdx.x * 256 + threadIdx.x) * 4;
    if (i + 3 < n) {
        float4 v = *(const float4*)(in + i);
        out[i + 0] = (half_t)v.x;
        out[i + 1] = (half_t)v.y;
        out[i + 2] = (half_t)v.z;
        out[i + 3] = (half_t)v.w;
    }
}

// ---------------------------------------------------------------------------
// Kernel 3: WMMA GEMM  C[m,n] = sum_k A[m,k] * W[n,k]   (A,W f16 row-major)
// Async-LDS double-buffered tiles: A 64x32 (shared by 4 waves), B 128x32
// (shared by 2 waves).  Each thread issues 3 async b128 copies per tile.
//   mode 0/1: store f16 to Qh/Kh [b][h][s][d]
//   mode 2:   store f16 to Vt    [b][h][d][s]
//   mode 3:   store f32  out[m][n] = acc + bias[n] + resid[m][n]
// ---------------------------------------------------------------------------
__global__ __launch_bounds__(256) void gemm_wmma(
    int mode, const half_t* __restrict__ A, const half_t* __restrict__ W,
    half_t* __restrict__ outH, const float* __restrict__ bias,
    const float* __restrict__ resid, float* __restrict__ outF) {
    __shared__ __align__(32) half_t Abuf[2][64 * 32];    // 4 KB x2
    __shared__ __align__(32) half_t Bbuf[2][128 * 32];   // 8 KB x2

    const int t    = threadIdx.x;
    const int lane = t & 31;
    const int wave = t >> 5;
    const int hi = lane >> 4, ln = lane & 15;
    const int lm  = (wave >> 2) * 32;          // wave row offset in macro-tile
    const int lnn = (wave & 3) * 32;           // wave col offset in macro-tile
    const int mBase = blockIdx.y * 64;
    const int nBase = blockIdx.x * 128;

    // cooperative copy indices (16B chunks)
    const int ar  = t >> 2,        ac  = (t & 3) * 8;          // A: 64 rows x 4
    const int br0 = t >> 2,        bc0 = (t & 3) * 8;          // B rows 0..63
    const int br1 = 64 + (t >> 2), bc1 = (t & 3) * 8;          // B rows 64..127

    auto issue = [&](int k0, int buf) {
        async_ld_b128(lds_off(&Abuf[buf][ar * 32 + ac]),
                      A + (size_t)(mBase + ar) * H_S + k0 + ac);
        async_ld_b128(lds_off(&Bbuf[buf][br0 * 32 + bc0]),
                      W + (size_t)(nBase + br0) * H_S + k0 + bc0);
        async_ld_b128(lds_off(&Bbuf[buf][br1 * 32 + bc1]),
                      W + (size_t)(nBase + br1) * H_S + k0 + bc1);
    };

    const int NK = H_S / 32;
    issue(0, 0);

    v8f acc[2][2] = {};
    for (int kt = 0; kt < NK; ++kt) {
        const int buf = kt & 1;
        if (kt + 1 < NK) {
            issue((kt + 1) * 32, buf ^ 1);
            WAIT_ASYNC(0x3);          // tile kt landed; next tile in flight
        } else {
            WAIT_ASYNC(0x0);
        }
        __syncthreads();

        v16h a0 = load_a_frag(&Abuf[buf][(lm + ln) * 32], 0, hi);
        v16h a1 = load_a_frag(&Abuf[buf][(lm + 16 + ln) * 32], 0, hi);
        v16h b0 = *(const v16h*)(&Bbuf[buf][(lnn + ln) * 32 + hi * 16]);
        v16h b1 = *(const v16h*)(&Bbuf[buf][(lnn + 16 + ln) * 32 + hi * 16]);
        acc[0][0] = wmma32(a0, b0, acc[0][0]);
        acc[0][1] = wmma32(a0, b1, acc[0][1]);
        acc[1][0] = wmma32(a1, b0, acc[1][0]);
        acc[1][1] = wmma32(a1, b1, acc[1][1]);

        __syncthreads();              // protect buffer reuse at kt+2's issue
    }

    const int m0 = mBase + lm, n0 = nBase + lnn;
#pragma unroll
    for (int i = 0; i < 2; i++)
#pragma unroll
        for (int j = 0; j < 2; j++)
#pragma unroll
            for (int r = 0; r < 8; r++) {
                const int m = m0 + i * 16 + r + 8 * hi;
                const int n = n0 + j * 16 + ln;
                const float val = acc[i][j][r];
                if (mode == 3) {
                    outF[(size_t)m * H_S + n] =
                        val + bias[n] + resid[(size_t)m * H_S + n];
                } else {
                    const int b = m >> 11, s = m & (SEQ - 1);
                    const int h = n >> 6,  d = n & (P_S - 1);
                    size_t idx;
                    if (mode == 2)  // V transposed: [b][h][d][s]
                        idx = (((size_t)(b * N_H + h) * P_S) + d) * SEQ + s;
                    else            // Q/K: [b][h][s][d]
                        idx = (((size_t)(b * N_H + h) * SEQ) + s) * P_S + d;
                    outH[idx] = (half_t)val;
                }
            }
}

// ---------------------------------------------------------------------------
// Kernel 4: flash attention per (b,h).  Wave = 16 query rows; the block's
// 8 waves share each 32-key K/V tile, staged once via async global->LDS
// (double-buffered; 2 async b128 copies per thread per tile).
// Per iteration: 4 score WMMAs + online softmax + 4 PV WMMAs.
// grid = (SEQ/128, BATCH*N_H), block = 256 (8 waves).
// ---------------------------------------------------------------------------
__global__ __launch_bounds__(256) void attn_wmma(
    const half_t* __restrict__ Qh, const half_t* __restrict__ Kh,
    const half_t* __restrict__ Vt, half_t* __restrict__ ctx) {
    __shared__ __align__(32) half_t lds_p[8][16 * 32];   // per-wave P tile, 8 KB
    __shared__ __align__(32) half_t Ktile[2][32 * 64];   // keys x d, 4 KB x2
    __shared__ __align__(32) half_t Vtile[2][64 * 32];   // d x keys, 4 KB x2

    const int t    = threadIdx.x;
    const int lane = t & 31;
    const int wave = t >> 5;
    const int hi = lane >> 4, ln = lane & 15;
    const int bh = blockIdx.y;               // b*16 + h
    const int q0 = blockIdx.x * 128 + wave * 16;

    const half_t* Q  = Qh + (size_t)bh * SEQ * P_S;
    const half_t* Kp = Kh + (size_t)bh * SEQ * P_S;
    const half_t* Vp = Vt + (size_t)bh * P_S * SEQ;

    // cooperative copy indices (16B chunks)
    const int kr = t >> 3, kc = (t & 7) * 8;   // K tile: 32 rows x 8 chunks
    const int vr = t >> 2, vc = (t & 3) * 8;   // V tile: 64 rows x 4 chunks

    auto issue = [&](int kb, int buf) {
        async_ld_b128(lds_off(&Ktile[buf][kr * 64 + kc]),
                      Kp + (size_t)(kb + kr) * P_S + kc);
        async_ld_b128(lds_off(&Vtile[buf][vr * 32 + vc]),
                      Vp + (size_t)vr * SEQ + kb + vc);
    };

    // Q fragments (row q0+ln, d = 0..31 and 32..63) — loop invariant
    const half_t* qrow = Q + (size_t)(q0 + ln) * P_S;
    const v16h qa0 = load_a_frag(qrow, 0, hi);
    const v16h qa1 = load_a_frag(qrow, 32, hi);

    v8f o0 = {}, o1 = {}, o2 = {}, o3 = {};
    float m_s[8], l_s[8];
#pragma unroll
    for (int r = 0; r < 8; r++) { m_s[r] = -3.0e38f; l_s[r] = 0.f; }

    half_t* pt = lds_p[wave];
    const int NT = SEQ / 32;
    issue(0, 0);

    for (int kt = 0; kt < NT; ++kt) {
        const int buf = kt & 1;
        if (kt + 1 < NT) {
            issue((kt + 1) * 32, buf ^ 1);
            WAIT_ASYNC(0x2);
        } else {
            WAIT_ASYNC(0x0);
        }
        __syncthreads();

        // ---- scores: S = Q * K^T (two 16-key tiles, K-dim 64 = 2x32)
        v8f s0 = {}, s1 = {};
        {
            const half_t* kr0 = &Ktile[buf][(size_t)ln * P_S];
            const half_t* kr1 = &Ktile[buf][(size_t)(16 + ln) * P_S];
            v16h b00 = *(const v16h*)(kr0 + hi * 16);
            v16h b01 = *(const v16h*)(kr0 + 32 + hi * 16);
            v16h b10 = *(const v16h*)(kr1 + hi * 16);
            v16h b11 = *(const v16h*)(kr1 + 32 + hi * 16);
            s0 = wmma32(qa0, b00, s0);
            s0 = wmma32(qa1, b01, s0);
            s1 = wmma32(qa0, b10, s1);
            s1 = wmma32(qa1, b11, s1);
        }

        // ---- online softmax update + stage P into LDS (f16)
#pragma unroll
        for (int r = 0; r < 8; r++) {
            const float a0 = s0[r] * 0.125f;   // 1/sqrt(64)
            const float a1 = s1[r] * 0.125f;
            float mx = rowmax16(fmaxf(a0, a1));
            const float mn = fmaxf(m_s[r], mx);
            const float alpha = __expf(m_s[r] - mn);
            m_s[r] = mn;
            const float p0 = __expf(a0 - mn);
            const float p1 = __expf(a1 - mn);
            l_s[r] = l_s[r] * alpha + rowsum16(p0 + p1);
            const int row = r + 8 * hi;
            pt[row * 32 + ln]      = (half_t)p0;
            pt[row * 32 + 16 + ln] = (half_t)p1;
            o0[r] *= alpha; o1[r] *= alpha; o2[r] *= alpha; o3[r] *= alpha;
        }

        // ---- reload P as A-fragment (16x32)
        v16h pa;
        {
            const half_t* pr = pt + ln * 32 + hi * 16;
            v8h lo = *(const v8h*)(pr);
            v8h hh = *(const v8h*)(pr + 8);
            pa = __builtin_shufflevector(lo, hh,
                                         0,1,2,3,4,5,6,7,8,9,10,11,12,13,14,15);
        }

        // ---- O += P * V  (V^T tile -> contiguous B-fragment LDS loads)
        v16h bv0 = *(const v16h*)(&Vtile[buf][(0 * 16 + ln) * 32 + hi * 16]);
        v16h bv1 = *(const v16h*)(&Vtile[buf][(1 * 16 + ln) * 32 + hi * 16]);
        v16h bv2 = *(const v16h*)(&Vtile[buf][(2 * 16 + ln) * 32 + hi * 16]);
        v16h bv3 = *(const v16h*)(&Vtile[buf][(3 * 16 + ln) * 32 + hi * 16]);
        o0 = wmma32(pa, bv0, o0);
        o1 = wmma32(pa, bv1, o1);
        o2 = wmma32(pa, bv2, o2);
        o3 = wmma32(pa, bv3, o3);

        __syncthreads();              // protect K/V buffer reuse
    }

    // ---- finalize: divide by l, write ctx row-major [b][s][H_S]
    const int b = bh >> 4, h = bh & 15;
#pragma unroll
    for (int r = 0; r < 8; r++) {
        const float invl = 1.0f / l_s[r];
        const int qidx = q0 + r + 8 * hi;
        const size_t base = ((size_t)b * SEQ + qidx) * H_S + h * P_S;
        ctx[base +  0 + ln] = (half_t)(o0[r] * invl);
        ctx[base + 16 + ln] = (half_t)(o1[r] * invl);
        ctx[base + 32 + ln] = (half_t)(o2[r] * invl);
        ctx[base + 48 + ln] = (half_t)(o3[r] * invl);
    }
}

// ---------------------------------------------------------------------------
// Host-side orchestration
// ---------------------------------------------------------------------------
extern "C" void kernel_launch(void* const* d_in, const int* in_sizes, int n_in,
                              void* d_out, int out_size, void* d_ws, size_t ws_size,
                              hipStream_t stream) {
    const float* q      = (const float*)d_in[0];
    const float* k      = (const float*)d_in[1];
    const float* v      = (const float*)d_in[2];
    const float* ln_q_g = (const float*)d_in[3];
    const float* ln_q_b = (const float*)d_in[4];
    const float* ln_k_g = (const float*)d_in[5];
    const float* ln_k_b = (const float*)d_in[6];
    const float* ln_v_g = (const float*)d_in[7];
    const float* ln_v_b = (const float*)d_in[8];
    const float* w_q    = (const float*)d_in[9];
    const float* w_k    = (const float*)d_in[10];
    const float* w_v    = (const float*)d_in[11];
    const float* w_fc   = (const float*)d_in[12];
    const float* b_fc   = (const float*)d_in[13];
    float* out = (float*)d_out;

    // f16 workspace layout
    half_t* ws = (half_t*)d_ws;
    const size_t MH = (size_t)MTOT * H_S;      // 8M halves
    const size_t WN = (size_t)H_S * H_S;       // 1M halves
    half_t* qn    = ws;
    half_t* kn    = qn + MH;
    half_t* vn    = kn + MH;
    half_t* wq16  = vn + MH;
    half_t* wk16  = wq16 + WN;
    half_t* wv16  = wk16 + WN;
    half_t* wfc16 = wv16 + WN;
    half_t* Qh    = wfc16 + WN;
    half_t* Kh    = Qh + MH;
    half_t* Vt    = Kh + MH;
    half_t* ctx   = Vt + MH;

    // 1) LayerNorm -> f16
    layernorm_to_f16<<<MTOT, 256, 0, stream>>>(q, ln_q_g, ln_q_b, qn);
    layernorm_to_f16<<<MTOT, 256, 0, stream>>>(k, ln_k_g, ln_k_b, kn);
    layernorm_to_f16<<<MTOT, 256, 0, stream>>>(v, ln_v_g, ln_v_b, vn);

    // 2) weights -> f16
    cvt_f32_to_f16<<<1024, 256, 0, stream>>>(w_q,  wq16,  (int)WN);
    cvt_f32_to_f16<<<1024, 256, 0, stream>>>(w_k,  wk16,  (int)WN);
    cvt_f32_to_f16<<<1024, 256, 0, stream>>>(w_v,  wv16,  (int)WN);
    cvt_f32_to_f16<<<1024, 256, 0, stream>>>(w_fc, wfc16, (int)WN);

    // 3) QKV projections (WMMA + async LDS staging)
    dim3 gg(H_S / 128, MTOT / 64);
    gemm_wmma<<<gg, 256, 0, stream>>>(0, qn, wq16, Qh, nullptr, nullptr, nullptr);
    gemm_wmma<<<gg, 256, 0, stream>>>(1, kn, wk16, Kh, nullptr, nullptr, nullptr);
    gemm_wmma<<<gg, 256, 0, stream>>>(2, vn, wv16, Vt, nullptr, nullptr, nullptr);

    // 4) flash attention (WMMA + async LDS staging of K/V)
    attn_wmma<<<dim3(SEQ / 128, BATCH * N_H), 256, 0, stream>>>(Qh, Kh, Vt, ctx);

    // 5) output projection + bias + residual (WMMA, fp32 out)
    gemm_wmma<<<gg, 256, 0, stream>>>(3, ctx, wfc16, nullptr, b_fc, q, out);
}
